// Dynamic_conv2d_6811818131749
// MI455X (gfx1250) — matrix operand order repaired and checked by reading, not compiled, not run on hardware
//
#include <hip/hip_runtime.h>
#include <hip/hip_bf16.h>

typedef __attribute__((ext_vector_type(16))) __bf16 v16bf;
typedef __attribute__((ext_vector_type(8)))  float  v8f;
typedef __attribute__((ext_vector_type(4)))  unsigned int u32x4;
typedef __attribute__((ext_vector_type(4)))  float  f32x4;

#define B_   16
#define CIN  128
#define T_   256
#define FR   128
#define COUT 128
#define NBK  4
#define HID  32
#define KW   3
#define KKT  (CIN * 9)      // 1152 = K dimension of implicit GEMM
#define M_   (NBK * COUT)   // 512  = M dimension (basis-major rows)

__device__ __forceinline__ unsigned short f2bf(float f) {
  unsigned u = __float_as_uint(f);
  u = u + 0x7FFFu + ((u >> 16) & 1u);   // round-to-nearest-even
  return (unsigned short)(u >> 16);
}

// ---------------- W f32 -> bf16 GEMM-A matrix [M_][KKT] ----------------
__global__ void wcvt_kernel(const float* __restrict__ W,
                            unsigned short* __restrict__ A) {
  int idx = blockIdx.x * 256 + threadIdx.x;
  if (idx >= M_ * KKT) return;
  int m  = idx / KKT;
  int kk = idx - m * KKT;
  int tap = kk >> 7;          // kk = tap*128 + ci
  int ci  = kk & 127;
  int k   = m >> 7;
  int co  = m & 127;
  int dy = tap / 3, dx = tap % 3;
  int src = (((k * COUT + co) * CIN + ci) * KW + dy) * KW + dx;
  A[idx] = f2bf(W[src]);
}

// ---------------- x f32 [b][ci][t][f] -> bf16 [b][t][f][ci] ----------------
// LDS-transposed so main-kernel B tiles are contiguous (ci innermost).
__global__ __launch_bounds__(256) void xcvt_kernel(const float* __restrict__ x,
                                                   unsigned short* __restrict__ xbf) {
  __shared__ __align__(16) unsigned short tile[128][72];  // [f][ci_local], padded
  const int t = blockIdx.x;
  const int b = blockIdx.y;
  const int tid = threadIdx.x;
  #pragma unroll
  for (int p = 0; p < 2; ++p) {            // two 64-ci halves
    __syncthreads();
    // load f32 coalesced along f, scatter bf16 into LDS transposed
    #pragma unroll
    for (int i = 0; i < 8; ++i) {          // 64ci x 128f = 2048 f32x4
      int lin = tid + i * 256;
      int cil = lin >> 5;                  // 0..63
      int fq  = lin & 31;                  // f quarter
      const f32x4* src = (const f32x4*)(x +
          (((size_t)(b * CIN + p * 64 + cil)) * T_ + t) * FR) + fq;
      f32x4 v = *src;
      tile[fq * 4 + 0][cil] = f2bf(v.x);
      tile[fq * 4 + 1][cil] = f2bf(v.y);
      tile[fq * 4 + 2][cil] = f2bf(v.z);
      tile[fq * 4 + 3][cil] = f2bf(v.w);
    }
    __syncthreads();
    // write bf16 coalesced along ci
    #pragma unroll
    for (int i = 0; i < 4; ++i) {          // 128f x 64ci / 8 = 1024 chunks
      int lin = tid + i * 256;
      int f     = lin >> 3;
      int chunk = lin & 7;
      u32x4 d = *((const u32x4*)&tile[f][chunk * 8]);
      *((u32x4*)(xbf + (((size_t)(b * T_ + t)) * FR + f) * CIN + p * 64 + chunk * 8)) = d;
    }
  }
}

// ---------------- mean over freq: xm[b,c,t] ----------------
__global__ __launch_bounds__(256) void meanf_kernel(const float* __restrict__ x,
                                                    float* __restrict__ xm) {
  int row  = (blockIdx.x * 256 + threadIdx.x) >> 5;   // one wave32 per row
  int lane = threadIdx.x & 31;
  const f32x4* p = (const f32x4*)(x + (size_t)row * FR);
  f32x4 v = p[lane];
  float s = v.x + v.y + v.z + v.w;
  #pragma unroll
  for (int off = 16; off; off >>= 1) s += __shfl_xor(s, off, 32);
  if (lane == 0) xm[row] = s * (1.0f / FR);
}

// ---------------- attention: conv1d(3)+BN+ReLU+conv1d(1)+softmax ----------------
__global__ __launch_bounds__(256) void attn_kernel(
    const float* __restrict__ xm, const float* __restrict__ w1,
    const float* __restrict__ gamma, const float* __restrict__ beta,
    const float* __restrict__ mean, const float* __restrict__ var,
    const float* __restrict__ w2, const float* __restrict__ b2,
    float* __restrict__ att) {
  __shared__ float sw1[HID * CIN * KW];       // 48 KB
  __shared__ float sscale[HID], sshift[HID], sw2[NBK * HID], sb2[NBK];
  int b = blockIdx.x, t = threadIdx.x;
  for (int i = t; i < HID * CIN * KW; i += 256) sw1[i] = w1[i];
  if (t < HID) {
    float sc = gamma[t] * rsqrtf(var[t] + 1e-5f);
    sscale[t] = sc;
    sshift[t] = beta[t] - mean[t] * sc;
  }
  if (t < NBK * HID) sw2[t] = w2[t];
  if (t < NBK) sb2[t] = b2[t];
  __syncthreads();

  float h[HID];
  #pragma unroll
  for (int j = 0; j < HID; j++) h[j] = 0.f;
  const float* xb = xm + (size_t)b * CIN * T_;
  for (int ci = 0; ci < CIN; ci++) {
    const float* row = xb + ci * T_;
    float x0 = (t > 0)      ? row[t - 1] : 0.f;
    float x1 = row[t];
    float x2 = (t < T_ - 1) ? row[t + 1] : 0.f;
    #pragma unroll
    for (int j = 0; j < HID; j++) {
      const float* wp = &sw1[(j * CIN + ci) * KW];
      h[j] = fmaf(wp[0], x0, fmaf(wp[1], x1, fmaf(wp[2], x2, h[j])));
    }
  }
  float a[NBK];
  #pragma unroll
  for (int k = 0; k < NBK; k++) a[k] = sb2[k];
  #pragma unroll
  for (int j = 0; j < HID; j++) {
    float hb = h[j] * sscale[j] + sshift[j];
    hb = hb > 0.f ? hb : 0.f;
    #pragma unroll
    for (int k = 0; k < NBK; k++) a[k] = fmaf(sw2[k * HID + j], hb, a[k]);
  }
  float mx = fmaxf(fmaxf(a[0], a[1]), fmaxf(a[2], a[3]));
  float e[NBK], se = 0.f;
  #pragma unroll
  for (int k = 0; k < NBK; k++) { e[k] = __expf((a[k] - mx) * (1.0f / 31.0f)); se += e[k]; }
  float inv = 1.f / se;
  float* ap = att + ((size_t)b * T_ + t) * NBK;
  #pragma unroll
  for (int k = 0; k < NBK; k++) ap[k] = e[k] * inv;
}

// ---------------- main implicit-GEMM dynamic conv with WMMA bf16 ----------------
union FragBF { u32x4 q[2]; v16bf v; };

__global__ __launch_bounds__(256, 1) void dynconv_kernel(
    const unsigned short* __restrict__ xbf, const unsigned short* __restrict__ A,
    const float* __restrict__ att, float* __restrict__ out) {
  __shared__ __align__(16) unsigned short Blds[FR][32];  // [f][K chunk], 8 KB
  const int t  = blockIdx.x;
  const int b  = blockIdx.y;
  const int tid  = threadIdx.x;
  const int lane = tid & 31;
  const int wave = tid >> 5;          // wave -> Cout slice wave*16..+15
  const int nrow = lane & 15;
  const int hsel = lane >> 4;

  v8f zero = {0, 0, 0, 0, 0, 0, 0, 0};
  u32x4 zero4 = {0, 0, 0, 0};
  v8f acc[NBK][8];                    // 4 bases x 8 column tiles (N=128)
  #pragma unroll
  for (int k = 0; k < NBK; k++)
    #pragma unroll
    for (int n = 0; n < 8; n++) acc[k][n] = zero;

  const int fRow = tid >> 1;          // staging: one f row per thread pair
  const int half = tid & 1;           // staging: 16-ci half (32B)

  for (int kc = 0; kc < KKT / 32; ++kc) {   // 36 K-steps
    int tap = kc >> 2;                       // 4 chunks of 32 ci per tap
    int ci0 = (kc & 3) * 32;
    int dy = tap / 3 - 1, dx = tap % 3 - 1;
    int ts = t + dy;
    bool tval = (unsigned)ts < (unsigned)T_;

    __syncthreads();
    {   // stage bf16 patch: async Global->LDS copy (no VGPR round-trip);
        // out-of-range rows get zero-filled through the DS pipe instead.
      int fs = fRow + dx;
      bool vld = tval && (unsigned)fs < (unsigned)FR;
      unsigned long long la64 =
          (unsigned long long)(size_t)&Blds[fRow][half * 16];
      unsigned ldsa = (unsigned)la64;   // LDS byte address for VDST
      if (vld) {
        unsigned long long ga = (unsigned long long)(size_t)(xbf +
            (((size_t)(b * T_ + ts)) * FR + fs) * CIN + ci0 + half * 16);
        // INST_OFFSET is added to BOTH the LDS and global address (ISA 10.x):
        asm volatile(
            "global_load_async_to_lds_b128 %0, %1, off\n\t"
            "global_load_async_to_lds_b128 %0, %1, off offset:16"
            :: "v"(ldsa), "v"(ga) : "memory");
      } else {
        u32x4* dst = (u32x4*)&Blds[fRow][0] + half * 2;
        dst[0] = zero4;
        dst[1] = zero4;
      }
    }
    // compiler flushes DScnt before the barrier; ASYNCcnt is ours to wait on
    asm volatile("s_wait_asynccnt 0" ::: "memory");
    __syncthreads();

    // A fragments: lane<16 holds K{0..15}, lane>=16 holds K{16..31}
    FragBF fa[NBK];
    const unsigned short* Abase = A + (size_t)kc * 32 + (size_t)hsel * 16;
    #pragma unroll
    for (int k = 0; k < NBK; k++) {
      int m = k * COUT + wave * 16 + nrow;
      const u32x4* p = (const u32x4*)(Abase + (size_t)m * KKT);
      fa[k].q[0] = p[0];          // K chunk lo
      fa[k].q[1] = p[1];          // K chunk +8
    }
    // B fragments: lanes 0-15 = K 0..15 of column n, lanes 16-31 = K 16..31
    FragBF fb[8];
    #pragma unroll
    for (int n = 0; n < 8; n++) {
      const u32x4* bp = (const u32x4*)&Blds[n * 16 + nrow][hsel * 16];
      fb[n].q[0] = bp[0];
      fb[n].q[1] = bp[1];
    }
    #pragma unroll
    for (int k = 0; k < NBK; k++)
      #pragma unroll
      for (int n = 0; n < 8; n++)
        acc[k][n] = __builtin_amdgcn_wmma_f32_16x16x32_bf16(
            false, fa[k].v, false, fb[n].v, (short)0, acc[k][n], false, false);

    if (kc + 1 < KKT / 32)   // hint next A chunk into cache
      __builtin_prefetch(Abase + 32 + (size_t)(wave * 16 + nrow) * KKT, 0, 1);
  }

  // epilogue: fold 4 basis accumulators with attention weights (uniform per block)
  f32x4 av = *(const f32x4*)(att + ((size_t)b * T_ + t) * NBK);
  #pragma unroll
  for (int n = 0; n < 8; n++) {
    #pragma unroll
    for (int r = 0; r < 8; r++) {
      int mco = wave * 16 + r + (hsel << 3);   // C layout: lane>=16 -> M+8
      float v = av.x * acc[0][n][r] + av.y * acc[1][n][r] +
                av.z * acc[2][n][r] + av.w * acc[3][n][r];
      out[(((size_t)b * COUT + mco) * T_ + t) * FR + n * 16 + nrow] = v;
    }
  }
}

extern "C" void kernel_launch(void* const* d_in, const int* in_sizes, int n_in,
                              void* d_out, int out_size, void* d_ws, size_t ws_size,
                              hipStream_t stream) {
  const float* x  = (const float*)d_in[0];
  const float* w1 = (const float*)d_in[1];
  const float* g  = (const float*)d_in[2];
  const float* be = (const float*)d_in[3];
  const float* mn = (const float*)d_in[4];
  const float* vr = (const float*)d_in[5];
  const float* w2 = (const float*)d_in[6];
  const float* b2 = (const float*)d_in[7];
  const float* W  = (const float*)d_in[8];
  float* out = (float*)d_out;

  char* ws = (char*)d_ws;
  unsigned short* A   = (unsigned short*)ws;                     // 1,179,648 B
  float* xm  = (float*)(ws + 1179648);                           // 2,097,152 B
  float* att = (float*)(ws + 3276800);                           //    65,536 B
  unsigned short* xbf = (unsigned short*)(ws + 3342336);         // 134,217,728 B

  wcvt_kernel<<<(M_ * KKT + 255) / 256, 256, 0, stream>>>(W, A);
  xcvt_kernel<<<dim3(T_, B_), 256, 0, stream>>>(x, xbf);
  meanf_kernel<<<(B_ * CIN * T_) / 8, 256, 0, stream>>>(x, xm);
  attn_kernel<<<B_, 256, 0, stream>>>(xm, w1, g, be, mn, vr, w2, b2, att);
  dynconv_kernel<<<dim3(T_, B_), 256, 0, stream>>>(xbf, A, att, out);
}
